// LeadLagAugmentedAttention_69939247448932
// MI455X (gfx1250) — compile-verified, matrix operand order checked by reading, not run on hardware
//
#include <hip/hip_runtime.h>

// ---------------------------------------------------------------------------
// LeadLag-augmented multi-head attention for MI455X (gfx1250, wave32, WMMA).
//
// B=16, N=1024, D=512, H=8, KD=64.  ~69 GFLOP vs ~200MB moved => compute
// bound on the matrix path: everything heavy runs through
// v_wmma_f32_16x16x32_bf16 (bf16 in, f32 accumulate).  Softmax is done
// flash-attention style (online max/sum) so NxN scores are never stored.
//
// Round-2 changes vs round-1 (driven by the emitted asm):
//  * GEMM waves widened to 32x64 output tiles (8 WMMAs / 12 b128 loads).
//  * All operand loads of a k-step issued into distinct registers BEFORE the
//    WMMA group, so the compiler can stagger s_wait_loadcnt instead of
//    draining to 0 before every WMMA.
//  * Attention kernel: K-tile / V-tile loads grouped ahead of their WMMAs,
//    bias loads hoisted to overlap with score WMMAs.
//  * V^T epilogue store packs row pairs into 32-bit stores.
// ---------------------------------------------------------------------------

typedef __bf16 bf16;
typedef __attribute__((ext_vector_type(2)))  __bf16 v2bf;
typedef __attribute__((ext_vector_type(8)))  __bf16 v8bf;
typedef __attribute__((ext_vector_type(16))) __bf16 v16bf;
typedef __attribute__((ext_vector_type(8)))  float  v8f;

#define B_    16
#define N_    1024
#define D_    512
#define H_    8
#define KD_   64
#define HD_   512            // H*KD
#define MTOT_ 16384          // B*N
#define SCALE_ 0.125f        // 1/sqrt(64)

// ---------------------------------------------------------------------------
// CDNA5 16-bit A/B operand layout for 16x16x32 WMMA (ISA 7.12.2):
//   lane L: row/col index = L%16, half h = L/16
//   VGPR 0..3 hold K = h*8 .. h*8+7       (one contiguous 16B chunk)
//   VGPR 4..7 hold K = 16+h*8 .. 16+h*8+7 (second contiguous 16B chunk)
// => loading a 16x32 tile from a row-major source is two b128 loads per lane.
// ---------------------------------------------------------------------------
__device__ __forceinline__ v16bf load_tile16x32(const bf16* rowptr, int hf) {
  v8bf lo = *(const v8bf*)(rowptr + hf * 8);
  v8bf hi = *(const v8bf*)(rowptr + 16 + hf * 8);
  return __builtin_shufflevector(lo, hi, 0, 1, 2, 3, 4, 5, 6, 7,
                                 8, 9, 10, 11, 12, 13, 14, 15);
}

__device__ __forceinline__ v8f wmma_bf16(v16bf a, v16bf b, v8f c) {
  // D = A(16x32) * B(32x16) + C, f32 accumulate.
  return __builtin_amdgcn_wmma_f32_16x16x32_bf16(
      /*neg_a=*/false, a, /*neg_b=*/false, b,
      /*c_mod=*/(short)0, c, /*reuse_a=*/false, /*reuse_b=*/false);
}

__device__ __forceinline__ v8f zero_v8f() {
  v8f z;
#pragma unroll
  for (int i = 0; i < 8; ++i) z[i] = 0.0f;
  return z;
}

// ---------------------------------------------------------------------------
// fp32 -> bf16 elementwise convert
// ---------------------------------------------------------------------------
__global__ void cvt_f32_bf16_kernel(const float* __restrict__ in,
                                    bf16* __restrict__ out, int n) {
  int i = blockIdx.x * blockDim.x + threadIdx.x;
  if (i < n) out[i] = (bf16)in[i];
}

// fp32 [512,512] -> transposed bf16 [512,512]: out[c][k] = in[k][c]
// (WMMA B-operand wants output-column rows contiguous over the K dim)
__global__ void cvt_transpose_w_kernel(const float* __restrict__ in,
                                       bf16* __restrict__ out) {
  int i = blockIdx.x * blockDim.x + threadIdx.x;  // 512*512 threads
  int c = i & (D_ - 1);
  int k = i >> 9;
  out[(size_t)c * D_ + k] = (bf16)in[i];
}

// ---------------------------------------------------------------------------
// 32x64 GEMM core: two A row-tiles share four B column-tiles.
// Loads are grouped ahead of the 8 WMMAs so load waits can stagger.
// ---------------------------------------------------------------------------
__device__ __forceinline__ void gemm32x64(const bf16* __restrict__ A, int lda,
                                          const bf16* __restrict__ Bt, int ldb,
                                          int row0, int col0, int Kdim,
                                          int M, int hf, v8f acc[2][4]) {
  const bf16* arow0 = A + (size_t)(row0 + M) * lda;
  const bf16* arow1 = A + (size_t)(row0 + 16 + M) * lda;
  for (int k0 = 0; k0 < Kdim; k0 += 32) {
    v16bf a0 = load_tile16x32(arow0 + k0, hf);
    v16bf a1 = load_tile16x32(arow1 + k0, hf);
    v16bf bt[4];
#pragma unroll
    for (int t = 0; t < 4; ++t)
      bt[t] = load_tile16x32(Bt + (size_t)(col0 + t * 16 + M) * ldb + k0, hf);
#pragma unroll
    for (int t = 0; t < 4; ++t) acc[0][t] = wmma_bf16(a0, bt[t], acc[0][t]);
#pragma unroll
    for (int t = 0; t < 4; ++t) acc[1][t] = wmma_bf16(a1, bt[t], acc[1][t]);
  }
}

// ---------------------------------------------------------------------------
// QKV projection: one wave computes a 32x64 tile of x @ W (K=512).
// grid.x = (16384/32)*(512/64) = 4096 tiles, grid.y = which (0=Q,1=K,2=V).
// Q is pre-scaled by 1/sqrt(KD) and stored [B,H,N,KD]; K stored [B,H,N,KD];
// V stored transposed [B,H,KD,N] so the P@V B-operand is row-contiguous.
// ---------------------------------------------------------------------------
__global__ __launch_bounds__(32) void qkv_gemm_kernel(
    const bf16* __restrict__ xb, const bf16* __restrict__ WqT,
    const bf16* __restrict__ WkT, const bf16* __restrict__ WvT,
    bf16* __restrict__ Qb, bf16* __restrict__ Kb, bf16* __restrict__ Vt) {
  const int which = blockIdx.y;
  const int tile = blockIdx.x;
  const int row0 = (tile >> 3) * 32;
  const int col0 = (tile & 7) * 64;
  const int lane = threadIdx.x;
  const int M = lane & 15, hf = lane >> 4;

  const bf16* Bt = (which == 0) ? WqT : ((which == 1) ? WkT : WvT);

  v8f acc[2][4];
#pragma unroll
  for (int rt = 0; rt < 2; ++rt)
#pragma unroll
    for (int t = 0; t < 4; ++t) acc[rt][t] = zero_v8f();

  gemm32x64(xb, D_, Bt, D_, row0, col0, D_, M, hf, acc);

  const int hh = col0 >> 6;  // 64 cols == exactly one head
  if (which == 2) {
    // V^T store: rows r, r+1 of a tile are adjacent n -> pack to b32 stores.
#pragma unroll
    for (int rt = 0; rt < 2; ++rt)
#pragma unroll
      for (int t = 0; t < 4; ++t)
#pragma unroll
        for (int r = 0; r < 8; r += 2) {
          int m = row0 + rt * 16 + r + 8 * hf;
          int bb = m >> 10;
          int n = m & (N_ - 1);
          int d = t * 16 + M;
          v2bf pk;
          pk[0] = (bf16)acc[rt][t][r];
          pk[1] = (bf16)acc[rt][t][r + 1];
          *(v2bf*)&Vt[(((size_t)(bb * H_ + hh)) * KD_ + d) * N_ + n] = pk;
        }
  } else {
    const float s = (which == 0) ? SCALE_ : 1.0f;
    bf16* dst = (which == 0) ? Qb : Kb;
#pragma unroll
    for (int rt = 0; rt < 2; ++rt)
#pragma unroll
      for (int t = 0; t < 4; ++t)
#pragma unroll
        for (int r = 0; r < 8; ++r) {
          int m = row0 + rt * 16 + r + 8 * hf;
          int bb = m >> 10;
          int n = m & (N_ - 1);
          int d = t * 16 + M;
          dst[(((size_t)(bb * H_ + hh)) * N_ + n) * KD_ + d] =
              (bf16)(acc[rt][t][r] * s);
        }
  }
}

// ---------------------------------------------------------------------------
// Flash attention: one wave per (b, h, 16-query tile).  grid = 16*8*64 = 8192.
// Per 32-key strip: 4 WMMAs for S, bias add, online softmax, LDS transpose of
// P, 4 WMMAs accumulating P@V into a 16x64 f32 accumulator.
// ---------------------------------------------------------------------------
__global__ __launch_bounds__(32) void attn_kernel(
    const bf16* __restrict__ Qb, const bf16* __restrict__ Kb,
    const bf16* __restrict__ Vt, const float* __restrict__ bias,
    const float* __restrict__ bscale, bf16* __restrict__ Attn) {
  const int wid = blockIdx.x;
  const int qt = wid & 63;
  const int bh = wid >> 6;
  const int hh = bh & (H_ - 1);
  const int bb = bh >> 3;
  const int q0 = qt * 16;
  const int lane = threadIdx.x;
  const int M = lane & 15, hf = lane >> 4;

  const float hscale = bscale[hh];
  const size_t headQK = (size_t)(bb * H_ + hh) * N_;   // rows of [N,KD]
  const size_t headV = (size_t)(bb * H_ + hh) * KD_;   // rows of [KD,N]

  // Q A-operand held in registers: 16 rows x 64 K -> two 16x32 chunks.
  const bf16* qrow = Qb + (headQK + q0 + M) * KD_;
  v16bf aQ0 = load_tile16x32(qrow, hf);
  v16bf aQ1 = load_tile16x32(qrow + 32, hf);

  float mrow[8], lrow[8];
  v8f o[4];
#pragma unroll
  for (int r = 0; r < 8; ++r) { mrow[r] = -1e30f; lrow[r] = 0.0f; }
#pragma unroll
  for (int t = 0; t < 4; ++t) o[t] = zero_v8f();

  __shared__ bf16 Plds[16][32];  // probability tile bounce buffer (1 KB)

  for (int kc = 0; kc < N_ / 32; ++kc) {
    const int n0 = kc * 32;

    // ---- load phase: bias values + K tiles for both 16-key sub-tiles ----
    float bv[2][8];
#pragma unroll
    for (int st = 0; st < 2; ++st) {
      const float* brow = bias + (size_t)(q0 + 8 * hf) * N_ + n0 + st * 16 + M;
#pragma unroll
      for (int r = 0; r < 8; ++r) bv[st][r] = brow[(size_t)r * N_];
    }
    v16bf kb[2][2];
#pragma unroll
    for (int st = 0; st < 2; ++st) {
      const bf16* krow = Kb + (headQK + n0 + st * 16 + M) * KD_;
      kb[st][0] = load_tile16x32(krow, hf);
      kb[st][1] = load_tile16x32(krow + 32, hf);
    }

    // ---- score WMMAs: S = Q @ K^T over KD=64 ----
    v8f s[2];
#pragma unroll
    for (int st = 0; st < 2; ++st) {
      v8f c = zero_v8f();
      c = wmma_bf16(aQ0, kb[st][0], c);
      c = wmma_bf16(aQ1, kb[st][1], c);
#pragma unroll
      for (int r = 0; r < 8; ++r) c[r] += hscale * bv[st][r];
      s[st] = c;
    }

    // ---- online softmax across the 32-wide strip ----
    // C layout: reg r holds row (r + 8*hf), each row spans one 16-lane group.
#pragma unroll
    for (int r = 0; r < 8; ++r) {
      float mx = fmaxf(s[0][r], s[1][r]);
#pragma unroll
      for (int off = 1; off < 16; off <<= 1)
        mx = fmaxf(mx, __shfl_xor(mx, off, 32));
      float mnew = fmaxf(mrow[r], mx);
      float sc = __expf(mrow[r] - mnew);
      mrow[r] = mnew;
      float p0 = __expf(s[0][r] - mnew);
      float p1 = __expf(s[1][r] - mnew);
      s[0][r] = p0;
      s[1][r] = p1;
      float rs = p0 + p1;
#pragma unroll
      for (int off = 1; off < 16; off <<= 1) rs += __shfl_xor(rs, off, 32);
      lrow[r] = lrow[r] * sc + rs;
#pragma unroll
      for (int t = 0; t < 4; ++t) o[t][r] *= sc;
    }

    // ---- transpose P (C layout -> A layout) through LDS ----
    __syncthreads();
#pragma unroll
    for (int st = 0; st < 2; ++st)
#pragma unroll
      for (int r = 0; r < 8; ++r)
        Plds[r + 8 * hf][st * 16 + M] = (bf16)s[st][r];
    __syncthreads();
    v16bf pa = load_tile16x32(&Plds[M][0], hf);

    // ---- O += P(16x32) @ V(32x64): load all V tiles, then 4 WMMAs ----
    v16bf vb[4];
#pragma unroll
    for (int t = 0; t < 4; ++t)
      vb[t] = load_tile16x32(Vt + (headV + t * 16 + M) * N_ + n0, hf);
#pragma unroll
    for (int t = 0; t < 4; ++t) o[t] = wmma_bf16(pa, vb[t], o[t]);
    __syncthreads();
  }

  // Normalize and store bf16 attention output [B*N, H*KD].
#pragma unroll
  for (int r = 0; r < 8; ++r) {
    float inv = 1.0f / lrow[r];
    int row = bb * N_ + q0 + r + 8 * hf;
#pragma unroll
    for (int t = 0; t < 4; ++t)
      Attn[(size_t)row * HD_ + hh * KD_ + t * 16 + M] = (bf16)(o[t][r] * inv);
  }
}

// ---------------------------------------------------------------------------
// Output projection: Attn[16384,512]_bf16 @ WoT -> f32 out [B,N,D].
// One wave per 32x64 tile, grid = 4096.
// ---------------------------------------------------------------------------
__global__ __launch_bounds__(32) void out_gemm_kernel(
    const bf16* __restrict__ Attn, const bf16* __restrict__ WoT,
    float* __restrict__ out) {
  const int tile = blockIdx.x;
  const int row0 = (tile >> 3) * 32;
  const int col0 = (tile & 7) * 64;
  const int lane = threadIdx.x;
  const int M = lane & 15, hf = lane >> 4;

  v8f acc[2][4];
#pragma unroll
  for (int rt = 0; rt < 2; ++rt)
#pragma unroll
    for (int t = 0; t < 4; ++t) acc[rt][t] = zero_v8f();

  gemm32x64(Attn, HD_, WoT, HD_, row0, col0, HD_, M, hf, acc);

#pragma unroll
  for (int rt = 0; rt < 2; ++rt)
#pragma unroll
    for (int t = 0; t < 4; ++t)
#pragma unroll
      for (int r = 0; r < 8; ++r)
        out[(size_t)(row0 + rt * 16 + r + 8 * hf) * D_ + col0 + t * 16 + M] =
            acc[rt][t][r];
}

// ---------------------------------------------------------------------------
// Host launcher.  Inputs: x, leadlag_bias, Wq, Wk, Wv, Wo, bias_scale.
// Workspace layout (bytes):
//   xb     : 16384*512*2  = 16.78 MB
//   WqT/WkT/WvT/WoT: 4 * 512*512*2 = 2.10 MB
//   Qb, Kb, Vt: 3 * 16.78 MB
//   AttnOut: 16.78 MB          -> ~86.5 MB total
// ---------------------------------------------------------------------------
extern "C" void kernel_launch(void* const* d_in, const int* in_sizes, int n_in,
                              void* d_out, int out_size, void* d_ws,
                              size_t ws_size, hipStream_t stream) {
  (void)in_sizes; (void)n_in; (void)out_size; (void)ws_size;

  const float* x = (const float*)d_in[0];
  const float* bias = (const float*)d_in[1];
  const float* Wq = (const float*)d_in[2];
  const float* Wk = (const float*)d_in[3];
  const float* Wv = (const float*)d_in[4];
  const float* Wo = (const float*)d_in[5];
  const float* bscale = (const float*)d_in[6];
  float* out = (float*)d_out;

  char* ws = (char*)d_ws;
  const size_t xbBytes = (size_t)MTOT_ * D_ * 2;
  const size_t wBytes = (size_t)D_ * HD_ * 2;
  const size_t headBytes = (size_t)B_ * H_ * N_ * KD_ * 2;

  bf16* xb = (bf16*)ws;               ws += xbBytes;
  bf16* WqT = (bf16*)ws;              ws += wBytes;
  bf16* WkT = (bf16*)ws;              ws += wBytes;
  bf16* WvT = (bf16*)ws;              ws += wBytes;
  bf16* WoT = (bf16*)ws;              ws += wBytes;
  bf16* Qb = (bf16*)ws;               ws += headBytes;
  bf16* Kb = (bf16*)ws;               ws += headBytes;
  bf16* Vt = (bf16*)ws;               ws += headBytes;
  bf16* AttnOut = (bf16*)ws;          ws += xbBytes;

  // 1) precision conversion (+ weight transpose for WMMA B-operands)
  {
    int n = MTOT_ * D_;
    cvt_f32_bf16_kernel<<<(n + 255) / 256, 256, 0, stream>>>(x, xb, n);
    cvt_transpose_w_kernel<<<(D_ * HD_) / 256, 256, 0, stream>>>(Wq, WqT);
    cvt_transpose_w_kernel<<<(D_ * HD_) / 256, 256, 0, stream>>>(Wk, WkT);
    cvt_transpose_w_kernel<<<(D_ * HD_) / 256, 256, 0, stream>>>(Wv, WvT);
    cvt_transpose_w_kernel<<<(D_ * HD_) / 256, 256, 0, stream>>>(Wo, WoT);
  }

  // 2) Q/K/V projections (4096 tiles x 3 matrices, one wave per 32x64 tile)
  qkv_gemm_kernel<<<dim3(4096, 3), 32, 0, stream>>>(xb, WqT, WkT, WvT, Qb, Kb,
                                                    Vt);

  // 3) flash attention with lead-lag bias (one wave per (b,h,q-tile))
  attn_kernel<<<B_ * H_ * (N_ / 16), 32, 0, stream>>>(Qb, Kb, Vt, bias, bscale,
                                                      AttnOut);

  // 4) output projection -> f32
  out_gemm_kernel<<<4096, 32, 0, stream>>>(AttnOut, WoT, out);
}
